// OutAggregate_30777735643291
// MI455X (gfx1250) — 8-bit matrix operand order repaired and checked by reading, compiled, not run on hardware
//
#include <hip/hip_runtime.h>
#include <stdint.h>

#define NQ   900
#define NP   960
#define NB   32
#define NC   91
#define MATB ((size_t)NP*NP)
#define T_B  0.9f
#define FP8_ONE 0x38

typedef int   v8i __attribute__((ext_vector_type(8)));
typedef float v8f __attribute__((ext_vector_type(8)));

#define LDS_STRIDE 80
#define TILE       192
#define SLAB_BYTES (TILE * LDS_STRIDE)
#define BUF_BYTES  (2 * SLAB_BYTES)

__global__ void build_adj_kernel(const float* __restrict__ bboxes,
                                 uint8_t* __restrict__ adj) {
    int j = blockIdx.x * 256 + threadIdx.x;
    int i = blockIdx.y;
    int b = blockIdx.z;
    if (j >= NP) return;
    uint8_t val = 0;
    if (i < NQ && j < NQ) {
        const float4* bb = (const float4*)bboxes;
        float4 bi = bb[(size_t)b*NQ + i];
        float4 bj = bb[(size_t)b*NQ + j];
        float ax0 = bi.x - 0.5f*bi.z, ay0 = bi.y - 0.5f*bi.w;
        float ax1 = bi.x + 0.5f*bi.z, ay1 = bi.y + 0.5f*bi.w;
        float bx0 = bj.x - 0.5f*bj.z, by0 = bj.y - 0.5f*bj.w;
        float bx1 = bj.x + 0.5f*bj.z, by1 = bj.y + 0.5f*bj.w;
        float areaA = (ax1-ax0)*(ay1-ay0);
        float areaB = (bx1-bx0)*(by1-by0);
        float iw = fminf(ax1,bx1) - fmaxf(ax0,bx0);
        float ih = fminf(ay1,by1) - fmaxf(ay0,by0);
        float inter = fmaxf(iw,0.0f) * fmaxf(ih,0.0f);
        float uni = areaA + areaB - inter;
        float iou = inter / uni;
        float ew = fmaxf(ax1,bx1) - fminf(ax0,bx0);
        float eh = fmaxf(ay1,by1) - fminf(ay0,by0);
        float areaE = fmaxf(ew,0.0f) * fmaxf(eh,0.0f);
        float giou = iou - (areaE - uni) / areaE;
        val = (giou > T_B) ? FP8_ONE : 0;
    }
    adj[(size_t)b*MATB + (size_t)i*NP + j] = val;
}

__global__ __launch_bounds__(256)
void closure_step_kernel(const uint8_t* __restrict__ Acur,
                         uint8_t* __restrict__ Anext) {
    __shared__ __align__(16) uint8_t lds[2 * BUF_BYTES];

    const int t    = threadIdx.x;
    const int lane = t & 31;
    const int wave = t >> 5;
    const int b    = blockIdx.z;
    const int rowbase = blockIdx.y * TILE;
    const int colbase = blockIdx.x * TILE;
    const size_t mb = (size_t)b * MATB;
    const uint8_t* A = Acur + mb;

    const int wr = wave & 1;
    const int wc = wave >> 1;
    const int m  = lane & 15;
    const int ca = (lane < 16) ? 0 : 8;
    const int cb = (lane < 16) ? 0 : 8;

    const uint32_t lds_base = (uint32_t)(uintptr_t)(&lds[0]);

    auto stage_async = [&](int buf, int k0) {
        #pragma unroll
        for (int q = 0; q < 6; ++q) {
            int slab = (q >= 3);
            int cc   = t + (q % 3) * 256;
            int r    = cc >> 2;
            int seg  = cc & 3;
            int grow = (slab ? colbase : rowbase) + r;
            uint64_t gaddr = (uint64_t)(uintptr_t)(A + (size_t)grow*NP + k0 + seg*16);
            uint32_t laddr = lds_base + buf*BUF_BYTES + slab*SLAB_BYTES
                           + r*LDS_STRIDE + seg*16;
            asm volatile("global_load_async_to_lds_b128 %0, %1, off"
                         :: "v"(laddr), "v"(gaddr) : "memory");
        }
    };

    auto loadA = [&](const uint8_t* base, int rt) -> v8i {
        const uint32_t* row = (const uint32_t*)(base + (rt*16 + m)*LDS_STRIDE + ca);
        v8i f;
        f[0]=row[0];  f[1]=row[1];  f[2]=row[4];  f[3]=row[5];
        f[4]=row[8];  f[5]=row[9];  f[6]=row[12]; f[7]=row[13];
        return f;
    };
    auto loadB = [&](const uint8_t* base, int qt) -> v8i {
        const uint32_t* row = (const uint32_t*)(base + SLAB_BYTES + (qt*16 + m)*LDS_STRIDE + cb);
        v8i f;
        f[0]=row[0];  f[1]=row[1];  f[2]=row[4];  f[3]=row[5];
        f[4]=row[8];  f[5]=row[9];  f[6]=row[12]; f[7]=row[13];
        return f;
    };

    v8f acc[6][3];
    #pragma unroll
    for (int ii = 0; ii < 6; ++ii)
        #pragma unroll
        for (int jj = 0; jj < 3; ++jj)
            acc[ii][jj] = (v8f){0,0,0,0,0,0,0,0};

    stage_async(0, 0);

    const int KSTEPS = NP / 64;
    #pragma unroll
    for (int ks = 0; ks < KSTEPS; ++ks) {
        const int cur = ks & 1;
        asm volatile("s_wait_asynccnt 0x0" ::: "memory");
        __syncthreads();

        if (ks + 1 < KSTEPS)
            stage_async(cur ^ 1, (ks + 1) * 64);

        const uint8_t* base = &lds[cur * BUF_BYTES];
        v8i af[6], bf[3];
        #pragma unroll
        for (int ii = 0; ii < 6; ++ii) af[ii] = loadA(base, wr*6 + ii);
        #pragma unroll
        for (int jj = 0; jj < 3; ++jj) bf[jj] = loadB(base, wc*3 + jj);

        #pragma unroll
        for (int ii = 0; ii < 6; ++ii)
            #pragma unroll
            for (int jj = 0; jj < 3; ++jj)
                acc[ii][jj] = __builtin_amdgcn_wmma_f32_16x16x64_fp8_fp8(
                    af[ii], bf[jj], (short)0, acc[ii][jj], false, false);
    }

    uint8_t* D = Anext + mb;
    const int n  = lane & 15;
    const int mo = (lane < 16) ? 0 : 8;
    #pragma unroll
    for (int ii = 0; ii < 6; ++ii) {
        #pragma unroll
        for (int jj = 0; jj < 3; ++jj) {
            int gr = rowbase + (wr*6 + ii)*16;
            int gc = colbase + (wc*3 + jj)*16 + n;
            #pragma unroll
            for (int vv = 0; vv < 8; ++vv)
                D[(size_t)(gr + mo + vv)*NP + gc] = (acc[ii][jj][vv] > 0.5f) ? FP8_ONE : 0;
        }
    }
}

__global__ void adj_to_f32_kernel(const uint8_t* __restrict__ adj,
                                  float* __restrict__ out) {
    const int QC = NQ / 4;
    int idx = blockIdx.x * 256 + threadIdx.x;
    int b   = blockIdx.y;
    if (idx >= NQ * QC) return;
    int i  = idx / QC;
    int j4 = idx - i * QC;
    uint32_t w = *(const uint32_t*)(adj + (size_t)b*MATB + (size_t)i*NP + j4*4);
    float4 r;
    r.x = (w & 0x000000FFu) ? 1.0f : 0.0f;
    r.y = (w & 0x0000FF00u) ? 1.0f : 0.0f;
    r.z = (w & 0x00FF0000u) ? 1.0f : 0.0f;
    r.w = (w & 0xFF000000u) ? 1.0f : 0.0f;
    ((float4*)(out + (size_t)b*NQ*NQ))[idx] = r;
}

__global__ __launch_bounds__(128)
void aggregate_kernel(const uint8_t* __restrict__ adj,
                      const float* __restrict__ bboxes,
                      float* __restrict__ out) {
    __shared__ float red[5][128];
    const int i = blockIdx.x;
    const int b = blockIdx.y;
    const int t = threadIdx.x;
    const uint8_t* row = adj + (size_t)b*MATB + (size_t)i*NP;
    const float4*  bb  = (const float4*)bboxes + (size_t)b*NQ;

    float cnt = 0.f, s0 = 0.f, s1 = 0.f, s2 = 0.f, s3 = 0.f;
    for (int j = t; j < NQ; j += 128) {
        if (row[j]) {
            float4 x = bb[j];
            cnt += 1.0f; s0 += x.x; s1 += x.y; s2 += x.z; s3 += x.w;
        }
    }
    red[0][t]=cnt; red[1][t]=s0; red[2][t]=s1; red[3][t]=s2; red[4][t]=s3;
    __syncthreads();
    for (int off = 64; off > 0; off >>= 1) {
        if (t < off) {
            #pragma unroll
            for (int k = 0; k < 5; ++k) red[k][t] += red[k][t + off];
        }
        __syncthreads();
    }
    if (t == 0) {
        float inv = 1.0f / (red[0][0] + 1e-6f);
        float4 r = { red[1][0]*inv, red[2][0]*inv, red[3][0]*inv, red[4][0]*inv };
        ((float4*)out)[(size_t)b*NQ + i] = r;
    }
}

extern "C" void kernel_launch(void* const* d_in, const int* in_sizes, int n_in,
                              void* d_out, int out_size, void* d_ws, size_t ws_size,
                              hipStream_t stream) {
    const float* bboxes = (const float*)d_in[0];
    const float* logits = (const float*)d_in[1];
    float* out = (float*)d_out;

    const size_t off_logits = (size_t)NB*NQ*4;
    const size_t off_adj    = off_logits + (size_t)NB*NQ*NC;

    uint8_t* buf0 = (uint8_t*)d_ws;
    uint8_t* buf1 = buf0 + (size_t)NB * MATB;

    build_adj_kernel<<<dim3((NP + 255)/256, NP, NB), 256, 0, stream>>>(bboxes, buf0);

    uint8_t* cur = buf0;
    uint8_t* nxt = buf1;
    for (int it = 0; it < 10; ++it) {
        closure_step_kernel<<<dim3(NP/TILE, NP/TILE, NB), 256, 0, stream>>>(cur, nxt);
        uint8_t* tmp = cur; cur = nxt; nxt = tmp;
    }

    adj_to_f32_kernel<<<dim3((NQ*(NQ/4) + 255)/256, NB), 256, 0, stream>>>(cur, out + off_adj);
    aggregate_kernel<<<dim3(NQ, NB), 128, 0, stream>>>(cur, bboxes, out);
    hipMemcpyAsync(out + off_logits, logits, (size_t)NB*NQ*NC*sizeof(float),
                   hipMemcpyDeviceToDevice, stream);
}
